// RoutedExperts_86311662780953
// MI455X (gfx1250) — matrix operand order repaired and checked by reading, not compiled, not run on hardware
//
#include <hip/hip_runtime.h>
#include <hip/hip_bf16.h>
#include <cstdint>

// ---------------- types ----------------
typedef __bf16        bf16x16 __attribute__((ext_vector_type(16)));
typedef float         f32x8   __attribute__((ext_vector_type(8)));
typedef float         f32x4   __attribute__((ext_vector_type(4)));
typedef int           i32x4   __attribute__((ext_vector_type(4)));
typedef unsigned int  u32x2   __attribute__((ext_vector_type(2)));

// ---------------- problem constants ----------------
constexpr int T_ = 4096;   // tokens
constexpr int D_ = 1024;   // model dim
constexpr int H_ = 2048;   // ffn hidden
constexpr int O_ = 1024;   // output dim
constexpr int E_ = 8;      // experts
constexpr int ROWS_ = T_ * 2;   // 8192 (token,expert) pairs

// ---------------- GEMM tiling ----------------
constexpr int BM = 128, BN = 128, BK = 32;
constexpr int LDST = 40;              // LDS halfword stride (80B, 16B aligned, bank-safe)
constexpr int NTHREADS = 256;         // 8 waves (wave32)
constexpr int MAXTILES = ROWS_ / BM;  // 64 worst-case M tiles per expert

// ---------------- async copy availability (probe via __has_builtin) ----------------
#if __has_builtin(__builtin_amdgcn_global_load_async_to_lds_b128) && \
    __has_builtin(__builtin_amdgcn_s_wait_asynccnt)
#define USE_ASYNC 1
#else
#define USE_ASYNC 0
#endif

#if USE_ASYNC
__device__ inline void async_cp16(const unsigned short* g, unsigned short* l) {
  // arg0: generic pointer to int4 (bit pattern == global address)
  // arg1: LDS (AS3) pointer = low 32 bits of the generic flat address
  __builtin_amdgcn_global_load_async_to_lds_b128(
      (i32x4*)g,
      (__attribute__((address_space(3))) i32x4*)(unsigned)(uintptr_t)l,
      0, 0);
}
#endif

// ---------------- helpers ----------------
__device__ inline unsigned short f2bf(float f) {
  unsigned int u = __float_as_uint(f);
  u += 0x7FFFu + ((u >> 16) & 1u);    // round-to-nearest-even
  return (unsigned short)(u >> 16);
}

__device__ inline void store4bf(unsigned short* dst, f32x4 v) {
  unsigned int lo = (unsigned int)f2bf(v.x) | ((unsigned int)f2bf(v.y) << 16);
  unsigned int hi = (unsigned int)f2bf(v.z) | ((unsigned int)f2bf(v.w) << 16);
  *(u32x2*)dst = (u32x2){lo, hi};
}

// A fragment (16x32 bf16, MxK): lane holds row M=lane%16.
// lanes 0-15:  elems 0-15 -> K0-15
// lanes 16-31: elems 0-15 -> K16-31
__device__ inline bf16x16 ld_fragA(const unsigned short* s, int row, int hi) {
  const unsigned short* p = s + row * LDST;
  union { i32x4 q[2]; bf16x16 v; } u;
  u.q[0] = *(const i32x4*)(p + hi * 16);
  u.q[1] = *(const i32x4*)(p + hi * 16 + 8);
  return u.v;
}

// B fragment (32x16 bf16, KxN), LDS holds B as [N][K]:
// lane holds col N=lane%16; elems j -> K = j + 16*(lane>=16)
__device__ inline bf16x16 ld_fragB(const unsigned short* s, int row, int hi) {
  const unsigned short* p = s + row * LDST + hi * 16;
  union { i32x4 q[2]; bf16x16 v; } u;
  u.q[0] = *(const i32x4*)(p);
  u.q[1] = *(const i32x4*)(p + 8);
  return u.v;
}

// ---------------- f32 -> bf16 bulk conversion ----------------
__global__ __launch_bounds__(NTHREADS)
void k_cvt(const float* __restrict__ src, unsigned short* __restrict__ dst, int n4) {
  int i = blockIdx.x * blockDim.x + threadIdx.x;
  if (i < n4) {
    f32x4 v = *(const f32x4*)(src + (size_t)i * 4);
    store4bf(dst + (size_t)i * 4, v);
  }
}

// ---------------- routing kernels ----------------
__global__ void k_zero_counts(int* counts) {
  if (threadIdx.x < 3 * E_) counts[threadIdx.x] = 0;  // counts, offsets, cursors
}

__global__ void k_count(const int* __restrict__ tki, int* __restrict__ counts) {
  int i = blockIdx.x * blockDim.x + threadIdx.x;
  if (i < ROWS_) atomicAdd(&counts[tki[i] & (E_ - 1)], 1);
}

__global__ void k_offsets(const int* __restrict__ counts, int* __restrict__ offsets) {
  if (threadIdx.x == 0) {
    int s = 0;
    for (int e = 0; e < E_; ++e) { offsets[e] = s; s += counts[e]; }
  }
}

__global__ void k_fill(const int* __restrict__ tki, const int* __restrict__ offsets,
                       int* __restrict__ cursors, int* __restrict__ tok_of_row,
                       int* __restrict__ pos_map) {
  int i = blockIdx.x * blockDim.x + threadIdx.x;
  if (i < ROWS_) {
    int e = tki[i] & (E_ - 1);
    int slot = atomicAdd(&cursors[e], 1);
    int r = offsets[e] + slot;
    tok_of_row[r] = i >> 1;   // token id
    pos_map[i] = r;           // (t,k) -> compact row
  }
}

// ---------------- kernel: gate_up GEMM + SiLU*up (bf16 WMMA, f32 acc) ----------------
__global__ __launch_bounds__(NTHREADS)
void k_gateup(const unsigned short* __restrict__ hbf, const unsigned short* __restrict__ gupbf,
              const int* __restrict__ counts, const int* __restrict__ offsets,
              const int* __restrict__ tok_of_row, unsigned short* __restrict__ hbuf) {
  const int e  = blockIdx.z;
  const int n0 = blockIdx.y * BN;   // column block within H
  const int m0 = blockIdx.x * BM;   // row block within this expert's token list
  const int cnt = counts[e];
  if (m0 >= cnt) return;
  const int offE = offsets[e];

  __shared__ unsigned short As[2][BM * LDST];
  __shared__ unsigned short Bg[2][BN * LDST];
  __shared__ unsigned short Bu[2][BN * LDST];

  const int tid = threadIdx.x;
  const int lane = tid & 31, l16 = lane & 15, hi = lane >> 4;
  const int wid = tid >> 5, wm = wid & 3, wn = wid >> 2;  // 4x2 wave grid, 32x64 per wave

  f32x8 accG[2][4], accU[2][4];
#pragma unroll
  for (int i = 0; i < 2; ++i)
#pragma unroll
    for (int f = 0; f < 4; ++f) { accG[i][f] = (f32x8)(0.0f); accU[i][f] = (f32x8)(0.0f); }

  // loader coords: 2 threads per row, 16 bf16 (32B) each
  const int lrow = tid >> 1;
  const int lk = (tid & 1) * 16;
  const int grow = m0 + lrow;
  const unsigned short* apt =
      (grow < cnt) ? hbf + (size_t)tok_of_row[offE + grow] * D_ + lk : nullptr;
  const unsigned short* gpt = gupbf + ((size_t)e * 2 * H_ + (n0 + lrow)) * D_ + lk;
  const unsigned short* upt = gupbf + ((size_t)e * 2 * H_ + (H_ + n0 + lrow)) * D_ + lk;
  const int dsoff = lrow * LDST + lk;
  constexpr int NK = D_ / BK;

#if USE_ASYNC
  // --- async global->LDS ping-pong, one barrier per k-step ---
  {
    if (apt) { async_cp16(apt, &As[0][dsoff]); async_cp16(apt + 8, &As[0][dsoff + 8]); }
    async_cp16(gpt, &Bg[0][dsoff]); async_cp16(gpt + 8, &Bg[0][dsoff + 8]);
    async_cp16(upt, &Bu[0][dsoff]); async_cp16(upt + 8, &Bu[0][dsoff + 8]);
  }
  for (int ki = 0; ki < NK; ++ki) {
    const int b = ki & 1;
    __builtin_amdgcn_s_wait_asynccnt(0);
    __syncthreads();
    if (ki + 1 < NK) {
      const int kb = (ki + 1) * BK, nb = b ^ 1;
      if (apt) { async_cp16(apt + kb, &As[nb][dsoff]); async_cp16(apt + kb + 8, &As[nb][dsoff + 8]); }
      async_cp16(gpt + kb, &Bg[nb][dsoff]); async_cp16(gpt + kb + 8, &Bg[nb][dsoff + 8]);
      async_cp16(upt + kb, &Bu[nb][dsoff]); async_cp16(upt + kb + 8, &Bu[nb][dsoff + 8]);
    }
#else
  // --- register-staged ping-pong fallback ---
  i32x4 ra0 = {0,0,0,0}, ra1 = {0,0,0,0}, rg0, rg1, ru0, ru1;
  if (apt) { ra0 = *(const i32x4*)(apt); ra1 = *(const i32x4*)(apt + 8); }
  rg0 = *(const i32x4*)(gpt); rg1 = *(const i32x4*)(gpt + 8);
  ru0 = *(const i32x4*)(upt); ru1 = *(const i32x4*)(upt + 8);
  for (int ki = 0; ki < NK; ++ki) {
    const int b = ki & 1;
    *(i32x4*)&As[b][dsoff] = ra0; *(i32x4*)&As[b][dsoff + 8] = ra1;
    *(i32x4*)&Bg[b][dsoff] = rg0; *(i32x4*)&Bg[b][dsoff + 8] = rg1;
    *(i32x4*)&Bu[b][dsoff] = ru0; *(i32x4*)&Bu[b][dsoff + 8] = ru1;
    __syncthreads();
    if (ki + 1 < NK) {
      const int kb = (ki + 1) * BK;
      if (apt) { ra0 = *(const i32x4*)(apt + kb); ra1 = *(const i32x4*)(apt + kb + 8); }
      rg0 = *(const i32x4*)(gpt + kb); rg1 = *(const i32x4*)(gpt + kb + 8);
      ru0 = *(const i32x4*)(upt + kb); ru1 = *(const i32x4*)(upt + kb + 8);
    }
#endif
    // ---- compute on buffer b ----
    bf16x16 af[2];
#pragma unroll
    for (int i = 0; i < 2; ++i) af[i] = ld_fragA(As[b], wm * 32 + i * 16 + l16, hi);
#pragma unroll
    for (int f = 0; f < 4; ++f) {
      bf16x16 bb = ld_fragB(Bg[b], wn * 64 + f * 16 + l16, hi);
#pragma unroll
      for (int i = 0; i < 2; ++i)
        accG[i][f] = __builtin_amdgcn_wmma_f32_16x16x32_bf16(
            false, af[i], false, bb, (short)0, accG[i][f], false, false);
    }
#pragma unroll
    for (int f = 0; f < 4; ++f) {
      bf16x16 bb = ld_fragB(Bu[b], wn * 64 + f * 16 + l16, hi);
#pragma unroll
      for (int i = 0; i < 2; ++i)
        accU[i][f] = __builtin_amdgcn_wmma_f32_16x16x32_bf16(
            false, af[i], false, bb, (short)0, accU[i][f], false, false);
    }
  }

  // epilogue: h = silu(g) * u, bf16 to workspace
#pragma unroll
  for (int i = 0; i < 2; ++i)
#pragma unroll
    for (int f = 0; f < 4; ++f)
#pragma unroll
      for (int v = 0; v < 8; ++v) {
        int lr = wm * 32 + i * 16 + hi * 8 + v;       // C layout: M = v + 8*hi
        if (m0 + lr < cnt) {
          float g = accG[i][f][v];
          float u = accU[i][f][v];
          float s = g / (1.0f + __expf(-g));          // silu
          int col = n0 + wn * 64 + f * 16 + l16;
          hbuf[(size_t)(offE + m0 + lr) * H_ + col] = f2bf(s * u);
        }
      }
}

// ---------------- kernel: down GEMM (bf16 WMMA, f32 acc) ----------------
__global__ __launch_bounds__(NTHREADS)
void k_down(const unsigned short* __restrict__ hbuf, const unsigned short* __restrict__ downbf,
            const int* __restrict__ counts, const int* __restrict__ offsets,
            float* __restrict__ h2) {
  const int e  = blockIdx.z;
  const int n0 = blockIdx.y * BN;   // column block within O
  const int m0 = blockIdx.x * BM;
  const int cnt = counts[e];
  if (m0 >= cnt) return;
  const int offE = offsets[e];

  __shared__ unsigned short As[2][BM * LDST];
  __shared__ unsigned short Bs[2][BN * LDST];

  const int tid = threadIdx.x;
  const int lane = tid & 31, l16 = lane & 15, hi = lane >> 4;
  const int wid = tid >> 5, wm = wid & 3, wn = wid >> 2;

  f32x8 acc[2][4];
#pragma unroll
  for (int i = 0; i < 2; ++i)
#pragma unroll
    for (int f = 0; f < 4; ++f) acc[i][f] = (f32x8)(0.0f);

  const int lrow = tid >> 1;
  const int lk = (tid & 1) * 16;
  const int grow = m0 + lrow;
  const unsigned short* apt =
      (grow < cnt) ? hbuf + (size_t)(offE + grow) * H_ + lk : nullptr;
  const unsigned short* bpt = downbf + ((size_t)e * O_ + (n0 + lrow)) * H_ + lk;
  const int dsoff = lrow * LDST + lk;
  constexpr int NK = H_ / BK;

#if USE_ASYNC
  {
    if (apt) { async_cp16(apt, &As[0][dsoff]); async_cp16(apt + 8, &As[0][dsoff + 8]); }
    async_cp16(bpt, &Bs[0][dsoff]); async_cp16(bpt + 8, &Bs[0][dsoff + 8]);
  }
  for (int ki = 0; ki < NK; ++ki) {
    const int b = ki & 1;
    __builtin_amdgcn_s_wait_asynccnt(0);
    __syncthreads();
    if (ki + 1 < NK) {
      const int kb = (ki + 1) * BK, nb = b ^ 1;
      if (apt) { async_cp16(apt + kb, &As[nb][dsoff]); async_cp16(apt + kb + 8, &As[nb][dsoff + 8]); }
      async_cp16(bpt + kb, &Bs[nb][dsoff]); async_cp16(bpt + kb + 8, &Bs[nb][dsoff + 8]);
    }
#else
  i32x4 ra0 = {0,0,0,0}, ra1 = {0,0,0,0}, rb0, rb1;
  if (apt) { ra0 = *(const i32x4*)(apt); ra1 = *(const i32x4*)(apt + 8); }
  rb0 = *(const i32x4*)(bpt); rb1 = *(const i32x4*)(bpt + 8);
  for (int ki = 0; ki < NK; ++ki) {
    const int b = ki & 1;
    *(i32x4*)&As[b][dsoff] = ra0; *(i32x4*)&As[b][dsoff + 8] = ra1;
    *(i32x4*)&Bs[b][dsoff] = rb0; *(i32x4*)&Bs[b][dsoff + 8] = rb1;
    __syncthreads();
    if (ki + 1 < NK) {
      const int kb = (ki + 1) * BK;
      if (apt) { ra0 = *(const i32x4*)(apt + kb); ra1 = *(const i32x4*)(apt + kb + 8); }
      rb0 = *(const i32x4*)(bpt + kb); rb1 = *(const i32x4*)(bpt + kb + 8);
    }
#endif
    bf16x16 af[2];
#pragma unroll
    for (int i = 0; i < 2; ++i) af[i] = ld_fragA(As[b], wm * 32 + i * 16 + l16, hi);
#pragma unroll
    for (int f = 0; f < 4; ++f) {
      bf16x16 bb = ld_fragB(Bs[b], wn * 64 + f * 16 + l16, hi);
#pragma unroll
      for (int i = 0; i < 2; ++i)
        acc[i][f] = __builtin_amdgcn_wmma_f32_16x16x32_bf16(
            false, af[i], false, bb, (short)0, acc[i][f], false, false);
    }
  }

#pragma unroll
  for (int i = 0; i < 2; ++i)
#pragma unroll
    for (int f = 0; f < 4; ++f)
#pragma unroll
      for (int v = 0; v < 8; ++v) {
        int lr = wm * 32 + i * 16 + hi * 8 + v;
        if (m0 + lr < cnt) {
          int col = n0 + wn * 64 + f * 16 + l16;
          h2[(size_t)(offE + m0 + lr) * O_ + col] = acc[i][f][v];
        }
      }
}

// ---------------- kernel: deterministic weighted combine ----------------
__global__ __launch_bounds__(NTHREADS)
void k_combine(const float* __restrict__ h2, const float* __restrict__ wts,
               const int* __restrict__ pos_map, float* __restrict__ out) {
  const int t = blockIdx.x;
  const int r0 = pos_map[2 * t], r1 = pos_map[2 * t + 1];
  const float w0 = wts[2 * t], w1 = wts[2 * t + 1];
  const int col = threadIdx.x * 4;                // 256 threads * 4 = 1024 = O_
  f32x4 a = *(const f32x4*)(h2 + (size_t)r0 * O_ + col);
  f32x4 b = *(const f32x4*)(h2 + (size_t)r1 * O_ + col);
  f32x4 r = w0 * a + w1 * b;
  *(f32x4*)(out + (size_t)t * O_ + col) = r;
}

// ---------------- launch ----------------
extern "C" void kernel_launch(void* const* d_in, const int* in_sizes, int n_in,
                              void* d_out, int out_size, void* d_ws, size_t ws_size,
                              hipStream_t stream) {
  const float* hidden = (const float*)d_in[0];
  const int*   tki    = (const int*)d_in[1];     // top_k_indices
  const float* tkw    = (const float*)d_in[2];   // top_k_weights
  const float* gup    = (const float*)d_in[3];   // [E, 2H, D]
  const float* down   = (const float*)d_in[4];   // [E, O, H]
  float* out = (float*)d_out;

  // workspace layout
  char* w = (char*)d_ws;
  int* counts  = (int*)w;          // E
  int* offsets = counts + E_;      // E
  int* cursors = counts + 2 * E_;  // E
  int* pos_map = counts + 4 * E_;  // ROWS_
  int* tok_row = pos_map + ROWS_;  // ROWS_
  size_t off = 65792;
  unsigned short* hbfA   = (unsigned short*)(w + off); off += (size_t)T_ * D_ * 2;        // 8 MB
  unsigned short* gupbf  = (unsigned short*)(w + off); off += (size_t)E_ * 2 * H_ * D_ * 2; // 64 MB
  unsigned short* downbf = (unsigned short*)(w + off); off += (size_t)E_ * O_ * H_ * 2;   // 32 MB
  unsigned short* hbuf   = (unsigned short*)(w + off); off += (size_t)ROWS_ * H_ * 2;     // 32 MB
  float*          h2     = (float*)(w + off);                                             // 32 MB

  // 0: one-time f32 -> bf16 conversions (weights stay resident in 192MB L2)
  int n4h = T_ * D_ / 4, n4g = E_ * 2 * H_ * D_ / 4, n4d = E_ * O_ * H_ / 4;
  k_cvt<<<(n4h + 255) / 256, NTHREADS, 0, stream>>>(hidden, hbfA, n4h);
  k_cvt<<<(n4g + 255) / 256, NTHREADS, 0, stream>>>(gup, gupbf, n4g);
  k_cvt<<<(n4d + 255) / 256, NTHREADS, 0, stream>>>(down, downbf, n4d);

  // 1-4: routing
  k_zero_counts<<<1, 64, 0, stream>>>(counts);
  k_count<<<(ROWS_ + 255) / 256, 256, 0, stream>>>(tki, counts);
  k_offsets<<<1, 32, 0, stream>>>(counts, offsets);
  k_fill<<<(ROWS_ + 255) / 256, 256, 0, stream>>>(tki, offsets, cursors, tok_row, pos_map);

  // 5: gate_up + silu*up  (grid: Mtiles x H/128 x E)
  k_gateup<<<dim3(MAXTILES, H_ / BN, E_), NTHREADS, 0, stream>>>(
      hbfA, gupbf, counts, offsets, tok_row, hbuf);

  // 6: down projection (grid: Mtiles x O/128 x E)
  k_down<<<dim3(MAXTILES, O_ / BN, E_), NTHREADS, 0, stream>>>(
      hbuf, downbf, counts, offsets, h2);

  // 7: weighted combine -> out
  k_combine<<<T_, NTHREADS, 0, stream>>>(h2, tkw, pos_map, out);
}
